// BeamSearchSampler_75333726372059
// MI455X (gfx1250) — matrix operand order repaired and checked by reading, not compiled, not run on hardware
//
#include <hip/hip_runtime.h>
#include <math.h>

// ---------------------------------------------------------------- constants
#define B_    32
#define K_    4
#define E_    8
#define V_    32000
#define D_    1024
#define L_    33          // seq buffer length
#define MAXS_ 32
#define PADT  0
#define BOST  2
#define EOST  3

#define MT_   8           // 128 rows / 16
#define KT_   32          // 1024 / 32
#define NT_   2000        // 32000 / 16
#define NPW_  4           // n-tiles per wave per pass

typedef unsigned short u16;
typedef __attribute__((ext_vector_type(16))) __bf16 v16bf;
typedef __attribute__((ext_vector_type(8)))  float  v8f;

// fp32 -> bf16 round-to-nearest-even
static __device__ __forceinline__ u16 f2bf(float f) {
    union { float f; unsigned u; } v; v.f = f;
    unsigned u = v.u;
    unsigned r = u + 0x7FFFu + ((u >> 16) & 1u);
    return (u16)(r >> 16);
}

// A-fragment element -> K index (contiguous: k = (lane&16) + e, matches B)
static __device__ __forceinline__ int kA(int lane, int e) {
    return (lane & 16) + e;
}

// ------------------------------------------------- kernel 1: convert W_head
// W [D,V] fp32 row-major -> Bfrag [NT][KT][32 lanes][16 bf16]
// B fragment: lane holds column n = lane&15, k = (lane&16) + e
__global__ void k_convert_w(const float* __restrict__ W, u16* __restrict__ Bf) {
    int tid  = blockIdx.x * 256 + threadIdx.x;
    int lane = tid & 31;
    int tile = tid >> 5;                 // tile = nt*KT_ + kt
    if (tile >= NT_ * KT_) return;
    int nt = tile >> 5;
    int kt = tile & 31;
    const float* src = W + (size_t)(kt * 32 + (lane & 16)) * V_ + nt * 16 + (lane & 15);
    unsigned w[8];
#pragma unroll
    for (int p = 0; p < 8; ++p) {
        unsigned lo = f2bf(src[(size_t)(2 * p) * V_]);
        unsigned hi = f2bf(src[(size_t)(2 * p + 1) * V_]);
        w[p] = lo | (hi << 16);
    }
    uint4* dst = (uint4*)(Bf + ((size_t)tile * 32 + lane) * 16);
    dst[0] = make_uint4(w[0], w[1], w[2], w[3]);
    dst[1] = make_uint4(w[4], w[5], w[6], w[7]);
}

// ------------------------------------------------- kernel 2: gather A frags
// Afrag [MT][KT][32 lanes][16 bf16]; row m = mt*16 + (lane&15) = b*K_+k
__global__ void k_gather_a(const float* __restrict__ emb, const int* __restrict__ seq,
                           int t, u16* __restrict__ Af) {
    int tid  = blockIdx.x * 256 + threadIdx.x;
    int lane = tid & 31;
    int tile = tid >> 5;                 // tile = mt*KT_ + kt
    if (tile >= MT_ * KT_) return;
    int mt = tile >> 5;
    int kt = tile & 31;
    int m   = mt * 16 + (lane & 15);
    int tok = seq[m * L_ + t];
    const float* er = emb + (size_t)tok * D_ + kt * 32;
    unsigned w[8];
#pragma unroll
    for (int p = 0; p < 8; ++p) {
        unsigned lo = f2bf(er[kA(lane, 2 * p)]);
        unsigned hi = f2bf(er[kA(lane, 2 * p + 1)]);
        w[p] = lo | (hi << 16);
    }
    uint4* dst = (uint4*)(Af + ((size_t)tile * 32 + lane) * 16);
    dst[0] = make_uint4(w[0], w[1], w[2], w[3]);
    dst[1] = make_uint4(w[4], w[5], w[6], w[7]);
}

// ------------------------------------------------- kernel 3: WMMA GEMM (pipelined)
// 8 waves/block: wave w -> m-tile w; block -> 4 n-tiles. 32 k-tile loop,
// register double-buffered: loads for kt+1 overlap the 4 WMMAs of kt.
__global__ void k_gemm(const u16* __restrict__ Af, const u16* __restrict__ Bf,
                       float* __restrict__ logits) {
    int lane = threadIdx.x & 31;
    int w    = threadIdx.x >> 5;         // m-tile
    int nt0  = blockIdx.x * NPW_;

    const u16* aPtr = Af + ((size_t)(w * KT_) * 32 + lane) * 16;            // +512/kt
    const u16* bPtr = Bf + ((size_t)(nt0 * KT_) * 32 + lane) * 16;          // +512/kt, +16384/j

    v8f acc[NPW_];
#pragma unroll
    for (int j = 0; j < NPW_; ++j)
        acc[j] = (v8f){0.f, 0.f, 0.f, 0.f, 0.f, 0.f, 0.f, 0.f};

    // prologue: fragments for kt = 0
    v16bf a_cur = *(const v16bf*)(aPtr);
    v16bf b_cur[NPW_];
#pragma unroll
    for (int j = 0; j < NPW_; ++j)
        b_cur[j] = *(const v16bf*)(bPtr + (size_t)j * 16384);

    for (int kt = 0; kt < KT_ - 1; ++kt) {
        // prefetch the B stream 4 k-tiles ahead (global_prefetch_b8)
        if (kt + 4 < KT_) {
#pragma unroll
            for (int j = 0; j < NPW_; ++j)
                __builtin_prefetch(bPtr + (size_t)j * 16384 + (size_t)(kt + 4) * 512, 0, 1);
        }
        // issue next iteration's loads before this iteration's WMMAs
        v16bf a_nxt = *(const v16bf*)(aPtr + (size_t)(kt + 1) * 512);
        v16bf b_nxt[NPW_];
#pragma unroll
        for (int j = 0; j < NPW_; ++j)
            b_nxt[j] = *(const v16bf*)(bPtr + (size_t)j * 16384 + (size_t)(kt + 1) * 512);

#pragma unroll
        for (int j = 0; j < NPW_; ++j)
            acc[j] = __builtin_amdgcn_wmma_f32_16x16x32_bf16(
                false, a_cur, false, b_cur[j], (short)0, acc[j], false, false);

        a_cur = a_nxt;
#pragma unroll
        for (int j = 0; j < NPW_; ++j) b_cur[j] = b_nxt[j];
    }
    // epilogue: last k-tile
#pragma unroll
    for (int j = 0; j < NPW_; ++j)
        acc[j] = __builtin_amdgcn_wmma_f32_16x16x32_bf16(
            false, a_cur, false, b_cur[j], (short)0, acc[j], false, false);

    // C/D layout: lane<16 -> M=r, lane>=16 -> M=r+8; N = lane&15
    int mBase = w * 16 + ((lane & 16) >> 1);
    int nLoc  = lane & 15;
#pragma unroll
    for (int j = 0; j < NPW_; ++j) {
        int n = (nt0 + j) * 16 + nLoc;
#pragma unroll
        for (int r = 0; r < 8; ++r)
            logits[(size_t)(mBase + r) * V_ + n] = acc[j][r];
    }
}

// ------------------------------------------------- kernel 4: per-row softmax stats + top-8
__global__ void k_rowtop(const float* __restrict__ logits,
                         float* __restrict__ top8v, int* __restrict__ top8i) {
    __shared__ float sv[256][8];
    __shared__ int   si[256][8];
    __shared__ float sm[256];
    __shared__ float ss[256];

    int row = blockIdx.x;
    int tid = threadIdx.x;
    const float* rp = logits + (size_t)row * V_;

    float m = -INFINITY, s = 0.f;
    float tv[8]; int ti[8];
#pragma unroll
    for (int j = 0; j < 8; ++j) { tv[j] = -INFINITY; ti[j] = 0x7fffffff; }

    for (int c = tid; c < V_; c += 256) {
        float x = rp[c];
        float mo = m;
        m = fmaxf(m, x);
        s = s * expf(mo - m) + expf(x - m);
        // register-resident sorted-insert (desc, tie -> lower index)
        float cv = x; int ci = c;
#pragma unroll
        for (int j = 0; j < 8; ++j) {
            bool gt = (cv > tv[j]) || (cv == tv[j] && ci < ti[j]);
            float nv = gt ? cv : tv[j]; int ni = gt ? ci : ti[j];
            cv = gt ? tv[j] : cv;       ci = gt ? ti[j] : ci;
            tv[j] = nv; ti[j] = ni;
        }
    }
#pragma unroll
    for (int j = 0; j < 8; ++j) { sv[tid][j] = tv[j]; si[tid][j] = ti[j]; }
    sm[tid] = m; ss[tid] = s;
    __syncthreads();

    for (int stride = 128; stride >= 1; stride >>= 1) {
        if (tid < stride) {
            float mv[8]; int mi[8];
            int ia = 0, ib = 0;
#pragma unroll
            for (int o = 0; o < 8; ++o) {
                float av = sv[tid][ia];          int ai = si[tid][ia];
                float bv = sv[tid + stride][ib]; int bi = si[tid + stride][ib];
                bool ta = (av > bv) || (av == bv && ai < bi);
                mv[o] = ta ? av : bv; mi[o] = ta ? ai : bi;
                if (ta) ++ia; else ++ib;
            }
#pragma unroll
            for (int o = 0; o < 8; ++o) { sv[tid][o] = mv[o]; si[tid][o] = mi[o]; }
            float ma = sm[tid], mb = sm[tid + stride];
            float m2 = fmaxf(ma, mb);
            ss[tid] = ss[tid] * expf(ma - m2) + ss[tid + stride] * expf(mb - m2);
            sm[tid] = m2;
        }
        __syncthreads();
    }
    if (tid == 0) {
        float lse = sm[0] + logf(ss[0]);
#pragma unroll
        for (int o = 0; o < 8; ++o) {
            top8v[row * 8 + o] = sv[0][o] - lse;   // logp
            top8i[row * 8 + o] = si[0][o];
        }
    }
}

// ------------------------------------------------- kernel 5a: step-0 expand
__global__ void k_step0(const float* __restrict__ top8v, const int* __restrict__ top8i,
                        int* __restrict__ seq, float* __restrict__ sc) {
    int b = blockIdx.x, e = threadIdx.x;
    if (e < K_) {
        seq[(b * K_ + e) * L_ + 1] = top8i[(b * K_ + 0) * 8 + e];
        sc [(b * K_ + e) * L_ + 1] = top8v[(b * K_ + 0) * 8 + e];
    }
}

// ------------------------------------------------- kernel 5: beam update step t
__global__ void k_beam_update(int t, const float* __restrict__ top8v,
                              const int* __restrict__ top8i,
                              int* __restrict__ seq, float* __restrict__ sc) {
    __shared__ float cval[32]; __shared__ int ctok[32];
    __shared__ float score[32];
    __shared__ int   sel[4];
    __shared__ int   nseq[4][L_]; __shared__ float nsc[4][L_];

    int b = blockIdx.x, e = threadIdx.x;      // 32 threads = K_*E_ candidates
    int k = e >> 3, ee = e & 7;
    const int*   srow  = seq + (b * K_ + k) * L_;
    const float* scrow = sc  + (b * K_ + k) * L_;

    int  last = srow[t];
    bool done = (last == PADT) || (last == EOST);
    float cv; int ci;
    if (done) { cv = (ee == 0) ? 0.0f : -INFINITY; ci = (ee == 0) ? PADT : ee; }
    else      { cv = top8v[(b * K_ + k) * 8 + ee]; ci = top8i[(b * K_ + k) * 8 + ee]; }

    float cum = 0.f; int len = 0;
    for (int j = 0; j < L_; ++j) { cum += scrow[j]; len += (srow[j] != PADT); }
    int hl = len + ((ci != PADT) ? 1 : 0);
    float lp = powf(5.0f + (float)hl, 0.6f) / powf(6.0f, 0.6f);
    score[e] = (cum + cv) / lp;
    cval[e] = cv; ctok[e] = ci;
    __syncthreads();

    if (e == 0) {   // top-4 of 32, tie -> lower flat index (JAX semantics)
        unsigned taken = 0u;
        for (int r = 0; r < K_; ++r) {
            int best = -1; float bv = 0.f;
            for (int c = 0; c < 32; ++c)
                if (!((taken >> c) & 1u) && (best < 0 || score[c] > bv)) { bv = score[c]; best = c; }
            taken |= 1u << best; sel[r] = best;
        }
    }
    __syncthreads();

    for (int idx = e; idx < K_ * L_; idx += 32) {     // stage parents in LDS
        int kn = idx / L_, pos = idx % L_;
        int parent = sel[kn] >> 3;
        nseq[kn][pos] = seq[(b * K_ + parent) * L_ + pos];
        nsc [kn][pos] = sc [(b * K_ + parent) * L_ + pos];
    }
    __syncthreads();

    if (e < K_) {
        int c = sel[e];
        nsc[e][t + 1] = cval[c];
        int prev = nseq[e][t];
        nseq[e][t + 1] = (prev == EOST || prev == PADT) ? PADT : ctok[c];
    }
    __syncthreads();

    for (int idx = e; idx < K_ * L_; idx += 32) {
        int kn = idx / L_, pos = idx % L_;
        seq[(b * K_ + kn) * L_ + pos] = nseq[kn][pos];
        sc [(b * K_ + kn) * L_ + pos] = nsc [kn][pos];
    }
}

// ------------------------------------------------- init / finalize
__global__ void k_init(int* __restrict__ seq, float* __restrict__ sc) {
    int tid = blockIdx.x * 256 + threadIdx.x;
    if (tid < B_ * K_ * L_) {
        seq[tid] = (tid % L_ == 0) ? BOST : PADT;
        sc[tid]  = 0.f;
    }
}

__global__ void k_finalize(const int* __restrict__ seq, const float* __restrict__ sc,
                           float* __restrict__ out) {
    int tid = blockIdx.x * 256 + threadIdx.x;
    const int NSEQ = B_ * K_ * L_;              // 4224
    if (tid < NSEQ) out[tid] = (float)seq[tid];
    else if (tid < 2 * NSEQ) out[tid] = sc[tid - NSEQ];
    else if (tid < 2 * NSEQ + B_ * K_) {
        int bk = tid - 2 * NSEQ;
        int cnt = 0;
        for (int j = 0; j < L_; ++j) cnt += (seq[bk * L_ + j] != PADT);
        out[tid] = (float)cnt;
    }
}

// ---------------------------------------------------------------- launcher
extern "C" void kernel_launch(void* const* d_in, const int* in_sizes, int n_in,
                              void* d_out, int out_size, void* d_ws, size_t ws_size,
                              hipStream_t stream) {
    const float* emb = (const float*)d_in[0];   // [V,D]
    const float* W   = (const float*)d_in[1];   // [D,V]
    float* out = (float*)d_out;
    char* ws = (char*)d_ws;

    // workspace layout (128B aligned)
    const size_t BF_OFF  = 0;                                   // 65,536,000 B
    const size_t AF_OFF  = BF_OFF + (size_t)NT_ * KT_ * 512 * 2;
    const size_t LG_OFF  = AF_OFF + (size_t)MT_ * KT_ * 512 * 2;
    const size_t T8V_OFF = LG_OFF + (size_t)B_ * K_ * V_ * 4;
    const size_t T8I_OFF = T8V_OFF + (size_t)B_ * K_ * E_ * 4;
    const size_t SEQ_OFF = T8I_OFF + (size_t)B_ * K_ * E_ * 4;
    const size_t SC_OFF  = SEQ_OFF + (size_t)B_ * K_ * L_ * 4 + 128;

    u16*   Bf     = (u16*)(ws + BF_OFF);
    u16*   Af     = (u16*)(ws + AF_OFF);
    float* logits = (float*)(ws + LG_OFF);
    float* t8v    = (float*)(ws + T8V_OFF);
    int*   t8i    = (int*)(ws + T8I_OFF);
    int*   seq    = (int*)(ws + SEQ_OFF);
    float* sc     = (float*)(ws + SC_OFF);

    // one-time per launch: bf16 fragment-layout copy of W_head + state init
    k_convert_w<<<(NT_ * KT_ * 32 + 255) / 256, 256, 0, stream>>>(W, Bf);
    k_init<<<(B_ * K_ * L_ + 255) / 256, 256, 0, stream>>>(seq, sc);

    for (int t = 0; t < MAXS_; ++t) {
        k_gather_a<<<(MT_ * KT_ * 32 + 255) / 256, 256, 0, stream>>>(emb, seq, t, Af);
        k_gemm<<<NT_ / NPW_, 256, 0, stream>>>(Af, Bf, logits);
        k_rowtop<<<B_ * K_, 256, 0, stream>>>(logits, t8v, t8i);
        if (t == 0) k_step0<<<B_, 32, 0, stream>>>(t8v, t8i, seq, sc);
        else        k_beam_update<<<B_, 32, 0, stream>>>(t, t8v, t8i, seq, sc);
    }
    k_finalize<<<(2 * B_ * K_ * L_ + B_ * K_ + 255) / 256, 256, 0, stream>>>(seq, sc, out);
}
